// Net_63728724738688
// MI455X (gfx1250) — compile-verified, matrix operand order checked by reading, not run on hardware
//
#include <hip/hip_runtime.h>
#include <hip/hip_bf16.h>

typedef __attribute__((ext_vector_type(16))) _Float16 v16h;
typedef __attribute__((ext_vector_type(8)))  _Float16 v8h;
typedef __attribute__((ext_vector_type(8)))  float    v8f;

#define BATCH 64
#define NNODE 1024
#define NEDGE 16384
#define KP1   820      // ceil(0.8*1024)
#define KP2   656      // ceil(0.8*820)
#define MPAD  832      // KP1 padded to multiple of 16 for WMMA tiles
#define F1    64
#define F2    128
#define MT_PER_BLK 13  // 52 M-tiles = 4 blocks x 13

// ---------------- workspace layout (bytes) ----------------
static constexpr size_t OFF_XKH   = 0;                                        // f16 B*MPAD*F1
static constexpr size_t OFF_W2H   = OFF_XKH   + (size_t)BATCH*MPAD*F1*2;      // f16 F2*F1 (N-major)
static constexpr size_t OFF_HH    = OFF_W2H   + (size_t)F2*F1*2;              // f32 B*MPAD*F2
static constexpr size_t OFF_H2    = OFF_HH    + (size_t)BATCH*MPAD*F2*4;      // f32 B*MPAD*F2
static constexpr size_t OFF_ELIST = OFF_H2    + (size_t)BATCH*MPAD*F2*4;      // i32 B*NEDGE
static constexpr size_t OFF_ROWP  = OFF_ELIST + (size_t)BATCH*NEDGE*4;        // i32 B*NNODE
static constexpr size_t OFF_DEG2  = OFF_ROWP  + (size_t)BATCH*NNODE*4;        // f32 B*MPAD
static constexpr size_t OFF_DIS2  = OFF_DEG2  + (size_t)BATCH*MPAD*4;         // f32 B*MPAD
static constexpr size_t OFF_SC2   = OFF_DIS2  + (size_t)BATCH*MPAD*4;         // f32 B*MPAD

// Bitonic sort of 1024 (score,idx) pairs in LDS.
// Order: descending score; ties -> ascending index (matches jax.lax.top_k).
__device__ __forceinline__ void bitonic_desc(float* s, int* id, int tid) {
  for (int k = 2; k <= NNODE; k <<= 1) {
    for (int j = k >> 1; j > 0; j >>= 1) {
      int ixj = tid ^ j;
      if (ixj > tid) {
        bool up = ((tid & k) == 0);
        float s1 = s[tid], s2 = s[ixj];
        int   i1 = id[tid], i2 = id[ixj];
        // "greater" = belongs later in desired order
        bool greater = (s1 < s2) || (s1 == s2 && i1 > i2);
        if (greater == up) { s[tid] = s2; s[ixj] = s1; id[tid] = i2; id[ixj] = i1; }
      }
      __syncthreads();
    }
  }
}

// ---------------- Kernel 1: GCN layer 1 (collapsed) + top-k pool 1 + CSR build ----------------
__global__ __launch_bounds__(1024)
void k1_gcn1_pool1(const float* __restrict__ x, const int* __restrict__ ei,
                   const float* __restrict__ W1, const float* __restrict__ b1,
                   const float* __restrict__ p1,
                   float* __restrict__ perm_out, _Float16* __restrict__ xkh,
                   int* __restrict__ elist, int* __restrict__ rowp,
                   float* __restrict__ deg2w, float* __restrict__ dis2w) {
  __shared__ float xs[NNODE], deg[NNODE], dis[NNODE], ssum[NNODE], av[NNODE], sc[NNODE];
  __shared__ int   idx[NNODE], nidx[NNODE], cnt[NNODE], cur[NNODE], scanb[NNODE];

  const int b = blockIdx.x, tid = threadIdx.x;
  const int* rowe = ei + (size_t)b * 2 * NEDGE;
  const int* cole = rowe + NEDGE;

  xs[tid]  = x[(size_t)b * NNODE + tid];
  deg[tid] = 1.0f;                       // segment_sum(+1) folded in
  __syncthreads();
  for (int e = tid; e < NEDGE; e += NNODE) atomicAdd(&deg[cole[e]], 1.0f);  // ds_add_f32
  __syncthreads();
  dis[tid]  = rsqrtf(deg[tid]);
  ssum[tid] = 0.0f;
  __syncthreads();
  for (int e = tid; e < NEDGE; e += NNODE) {
    int r = rowe[e];
    atomicAdd(&ssum[cole[e]], xs[r] * dis[r]);                              // ds_add_f32
  }
  __syncthreads();

  // a[c] collapses gcn1: h1[c][f] = relu(a[c]*W1[f] + b1[f])
  float a = ssum[tid] * dis[tid] + xs[tid] / deg[tid];
  av[tid] = a;

  float np = 0.f;
  #pragma unroll 8
  for (int f = 0; f < F1; ++f) np += p1[f] * p1[f];
  float dot = 0.f;
  #pragma unroll 8
  for (int f = 0; f < F1; ++f) {
    float h = fmaxf(a * W1[f] + b1[f], 0.f);
    dot += h * p1[f];
  }
  sc[tid]  = tanhf(dot / sqrtf(np));
  idx[tid] = tid;
  __syncthreads();

  bitonic_desc(sc, idx, tid);

  nidx[tid] = -1;
  __syncthreads();
  const int   pm = idx[tid];
  const float sv = sc[tid];
  if (tid < KP1) {
    nidx[pm] = tid;
    perm_out[(size_t)b * KP1 + tid] = (float)pm;   // tuple output #2 (as f32)
  }
  __syncthreads();

  // xk = h1[perm] * sv, emitted directly in f16, padded to MPAD rows
  if (tid < MPAD) {
    _Float16* dst = xkh + ((size_t)b * MPAD + tid) * F1;
    if (tid < KP1) {
      float aa = av[pm];
      #pragma unroll 8
      for (int f = 0; f < F1; ++f)
        dst[f] = (_Float16)(fmaxf(aa * W1[f] + b1[f], 0.f) * sv);
    } else {
      #pragma unroll 8
      for (int f = 0; f < F1; ++f) dst[f] = (_Float16)0.f;
    }
  }

  // CSR build over kept, remapped edges (nr,nc) with nc as segment key
  cnt[tid] = 0; cur[tid] = 0;
  __syncthreads();
  for (int e = tid; e < NEDGE; e += NNODE) {
    int nr = nidx[rowe[e]], nc = nidx[cole[e]];
    if (nr >= 0 && nc >= 0) atomicAdd(&cnt[nc], 1);                         // ds_add_u32
  }
  __syncthreads();
  if (tid < MPAD) {
    float d2 = ((tid < KP1) ? (float)cnt[tid] : 0.f) + 1.0f;
    deg2w[(size_t)b * MPAD + tid] = d2;
    dis2w[(size_t)b * MPAD + tid] = rsqrtf(d2);
  }
  // Hillis-Steele inclusive scan -> exclusive base
  int v = cnt[tid];
  scanb[tid] = v;
  __syncthreads();
  for (int off = 1; off < NNODE; off <<= 1) {
    int t = (tid >= off) ? scanb[tid - off] : 0;
    __syncthreads();
    scanb[tid] += t;
    __syncthreads();
  }
  int base = scanb[tid] - v;
  rowp[(size_t)b * NNODE + tid] = base;
  cnt[tid] = base;                       // reuse cnt as exclusive base
  __syncthreads();
  for (int e = tid; e < NEDGE; e += NNODE) {
    int nr = nidx[rowe[e]], nc = nidx[cole[e]];
    if (nr >= 0 && nc >= 0) {
      int pos = cnt[nc] + atomicAdd(&cur[nc], 1);                           // ds_add_rtn_u32
      elist[(size_t)b * NEDGE + pos] = nr;
    }
  }
}

// ---------------- Kernel 2: W2 (K-major 64x128) -> f16 N-major (128x64) for WMMA B ----------------
__global__ void k2_w2(const float* __restrict__ W2, _Float16* __restrict__ w2h) {
  for (int i = threadIdx.x; i < F2 * F1; i += blockDim.x) {
    int n = i >> 6, k = i & 63;
    w2h[i] = (_Float16)W2[k * F2 + n];
  }
}

// ---------------- Kernel 3: hh = xk @ W2 via v_wmma_f32_16x16x32_f16 ----------------
// grid = (4, BATCH), block = 256 (8 waves); wave w owns N-tile w and streams 13 M-tiles
// against a register-resident B tile (W2 is constant across M and batch).
__global__ __launch_bounds__(256)
void k3_gemm(const _Float16* __restrict__ xkh, const _Float16* __restrict__ w2h,
             float* __restrict__ hh) {
  const int b = blockIdx.y;
  const int wave = threadIdx.x >> 5, lane = threadIdx.x & 31;
  const int kg = lane >> 4, mr = lane & 15;        // lane = 16*kg + (row-within-tile)
  const int n0 = wave * 16;

  // B tile: column n = n0+mr of W2 (stored N-major), K chunks per WMMA layout.
  const _Float16* Br = w2h + (size_t)(n0 + mr) * F1;
  v16h B0, B1;
  {
    v8h blo0 = *(const v8h*)(Br + 8 * kg);
    v8h bhi0 = *(const v8h*)(Br + 8 * kg + 16);
    v8h blo1 = *(const v8h*)(Br + 32 + 8 * kg);
    v8h bhi1 = *(const v8h*)(Br + 32 + 8 * kg + 16);
    #pragma unroll
    for (int i = 0; i < 8; ++i) {
      B0[i] = blo0[i]; B0[8 + i] = bhi0[i];
      B1[i] = blo1[i]; B1[8 + i] = bhi1[i];
    }
  }

  #pragma unroll 4
  for (int t = 0; t < MT_PER_BLK; ++t) {
    const int m0 = (blockIdx.x * MT_PER_BLK + t) * 16;
    const _Float16* Ar = xkh + ((size_t)b * MPAD + m0 + mr) * F1;
    v8h alo0 = *(const v8h*)(Ar + 8 * kg);
    v8h ahi0 = *(const v8h*)(Ar + 8 * kg + 16);
    v8h alo1 = *(const v8h*)(Ar + 32 + 8 * kg);
    v8h ahi1 = *(const v8h*)(Ar + 32 + 8 * kg + 16);
    v16h A0, A1;
    #pragma unroll
    for (int i = 0; i < 8; ++i) {
      A0[i] = alo0[i]; A0[8 + i] = ahi0[i];
      A1[i] = alo1[i]; A1[8 + i] = ahi1[i];
    }
    v8f c = {};
    c = __builtin_amdgcn_wmma_f32_16x16x32_f16(false, A0, false, B0, (short)0, c, false, false);
    c = __builtin_amdgcn_wmma_f32_16x16x32_f16(false, A1, false, B1, (short)0, c, false, false);

    // D layout: lane n = lane&15; VGPR r -> row r (lanes 0-15) or 8+r (lanes 16-31)
    float* D = hh + ((size_t)b * MPAD + m0 + ((lane < 16) ? 0 : 8)) * F2 + n0 + mr;
    #pragma unroll
    for (int r = 0; r < 8; ++r) D[(size_t)r * F2] = c[r];
  }
}

// ---------------- Kernel 4: GCN layer 2 aggregation (CSR gather) + relu + pool-2 scores ----------------
__global__ __launch_bounds__(256)
void k4_agg(const float* __restrict__ hh, const int* __restrict__ elist,
            const int* __restrict__ rowp, const float* __restrict__ deg2,
            const float* __restrict__ dis2, const float* __restrict__ b2,
            const float* __restrict__ p2, float* __restrict__ h2,
            float* __restrict__ score2) {
  const int b = blockIdx.x;
  const int wave = threadIdx.x >> 5, lane = threadIdx.x & 31;

  float np2 = 0.f;
  #pragma unroll 8
  for (int f = 0; f < F2; ++f) np2 += p2[f] * p2[f];
  const float inv_np2 = 1.f / sqrtf(np2);

  const int*   rp   = rowp  + (size_t)b * NNODE;
  const int*   el   = elist + (size_t)b * NEDGE;
  const float* d2v  = deg2  + (size_t)b * MPAD;
  const float* disv = dis2  + (size_t)b * MPAD;
  const float* hhb  = hh    + (size_t)b * MPAD * F2;

  for (int c = wave; c < KP1; c += 8) {
    float acc0 = 0.f, acc1 = 0.f, acc2 = 0.f, acc3 = 0.f;
    const int e0 = rp[c], e1 = rp[c + 1];
    for (int e = e0; e < e1; ++e) {
      int r = el[e];
      if (e + 1 < e1) {
        // overlap next gather's L2 latency: emits global_prefetch_b8
        __builtin_prefetch(hhb + (size_t)el[e + 1] * F2 + lane, 0, 0);
      }
      float dr = disv[r];
      const float* hr = hhb + (size_t)r * F2;
      acc0 += hr[lane     ] * dr;
      acc1 += hr[lane + 32] * dr;
      acc2 += hr[lane + 64] * dr;
      acc3 += hr[lane + 96] * dr;
    }
    const float dc = disv[c], dd = d2v[c];
    const float* hc = hhb + (size_t)c * F2;
    float*       ho = h2 + ((size_t)b * MPAD + c) * F2;
    float vals[4] = {acc0, acc1, acc2, acc3};
    float sdot = 0.f;
    #pragma unroll
    for (int q = 0; q < 4; ++q) {
      int f = lane + 32 * q;
      float vv = fmaxf(vals[q] * dc + hc[f] / dd + b2[f], 0.f);
      ho[f] = vv;
      sdot += vv * p2[f];
    }
    #pragma unroll
    for (int o = 16; o > 0; o >>= 1) sdot += __shfl_down(sdot, o, 32);   // wave32 reduce
    if (lane == 0) score2[(size_t)b * MPAD + c] = tanhf(sdot * inv_np2);
  }
}

// ---------------- Kernel 5: top-k pool 2 -> global max -> MLP head -> log_softmax ----------------
__global__ __launch_bounds__(1024)
void k5_head(const float* __restrict__ h2, const float* __restrict__ score2,
             const float* __restrict__ W3, const float* __restrict__ b3,
             const float* __restrict__ W4, const float* __restrict__ b4,
             float* __restrict__ out) {
  __shared__ float sc[NNODE];
  __shared__ int   id2[NNODE];
  __shared__ float g[F2];
  __shared__ float tbuf[64];
  const int b = blockIdx.x, tid = threadIdx.x;

  sc[tid]  = (tid < KP1) ? score2[(size_t)b * MPAD + tid] : -3.0e38f;
  id2[tid] = tid;
  __syncthreads();
  bitonic_desc(sc, id2, tid);

  if (tid < F2) {
    float m = -3.0e38f;
    for (int i = 0; i < KP2; ++i) {
      int nd = id2[i];
      m = fmaxf(m, h2[((size_t)b * MPAD + nd) * F2 + tid] * sc[i]);
    }
    g[tid] = m;
  }
  __syncthreads();
  if (tid < 64) {
    float acc = b3[tid];
    #pragma unroll 8
    for (int f = 0; f < F2; ++f) acc += g[f] * W3[f * 64 + tid];
    tbuf[tid] = fmaxf(acc, 0.f);
  }
  __syncthreads();
  if (tid == 0) {
    float l0 = b4[0], l1 = b4[1];
    #pragma unroll 8
    for (int j = 0; j < 64; ++j) { l0 += tbuf[j] * W4[j * 2]; l1 += tbuf[j] * W4[j * 2 + 1]; }
    float mx  = fmaxf(l0, l1);
    float lse = mx + logf(expf(l0 - mx) + expf(l1 - mx));
    out[(size_t)b * 2]     = l0 - lse;
    out[(size_t)b * 2 + 1] = l1 - lse;
  }
}

// ---------------- host launcher ----------------
extern "C" void kernel_launch(void* const* d_in, const int* in_sizes, int n_in,
                              void* d_out, int out_size, void* d_ws, size_t ws_size,
                              hipStream_t stream) {
  const float* x  = (const float*)d_in[0];
  const int*   ei = (const int*)  d_in[1];
  const float* W1 = (const float*)d_in[2];
  const float* b1 = (const float*)d_in[3];
  const float* p1 = (const float*)d_in[4];
  const float* W2 = (const float*)d_in[5];
  const float* b2 = (const float*)d_in[6];
  const float* p2 = (const float*)d_in[7];
  const float* W3 = (const float*)d_in[8];
  const float* b3 = (const float*)d_in[9];
  const float* W4 = (const float*)d_in[10];
  const float* b4 = (const float*)d_in[11];

  char* ws = (char*)d_ws;
  _Float16* xkh   = (_Float16*)(ws + OFF_XKH);
  _Float16* w2h   = (_Float16*)(ws + OFF_W2H);
  float*    hh    = (float*)   (ws + OFF_HH);
  float*    h2    = (float*)   (ws + OFF_H2);
  int*      elist = (int*)     (ws + OFF_ELIST);
  int*      rowp  = (int*)     (ws + OFF_ROWP);
  float*    deg2  = (float*)   (ws + OFF_DEG2);
  float*    dis2  = (float*)   (ws + OFF_DIS2);
  float*    sc2   = (float*)   (ws + OFF_SC2);

  float* out      = (float*)d_out;
  float* perm_out = out + BATCH * 2;   // tuple: [log_softmax (B,2)] then [perm (B,K1)]

  k1_gcn1_pool1<<<BATCH, NNODE, 0, stream>>>(x, ei, W1, b1, p1,
                                             perm_out, xkh, elist, rowp, deg2, dis2);
  k2_w2<<<1, 256, 0, stream>>>(W2, w2h);
  k3_gemm<<<dim3(MPAD / 16 / MT_PER_BLK, BATCH), 256, 0, stream>>>(xkh, w2h, hh);
  k4_agg<<<BATCH, 256, 0, stream>>>(hh, elist, rowp, deg2, dis2, b2, p2, h2, sc2);
  k5_head<<<BATCH, NNODE, 0, stream>>>(h2, sc2, W3, b3, W4, b4, out);
}
